// CausalSelfAttention_15942918603039
// MI455X (gfx1250) — matrix operand order repaired and checked by reading, not compiled, not run on hardware
//
#include <hip/hip_runtime.h>

typedef __attribute__((ext_vector_type(16))) _Float16     v16h;
typedef __attribute__((ext_vector_type(8)))  float        v8f;
typedef __attribute__((ext_vector_type(4)))  unsigned int v4u;
typedef __attribute__((ext_vector_type(8)))  int          v8i;
typedef __attribute__((ext_vector_type(4)))  int          v4i;

// ---------------------------------------------------------------------------
// WMMA helpers (v_wmma_f32_16x16x32_f16, wave32)
// ---------------------------------------------------------------------------
__device__ __forceinline__ v8f wmma32(v16h a, v16h b, v8f c) {
  return __builtin_amdgcn_wmma_f32_16x16x32_f16(false, a, false, b, (short)0, c,
                                                false, false);
}

// A-operand (16x32 f16): lane m = lane%16, g = lane/16.
// halves hi = A[m][16g + hi]  (32B contiguous)
__device__ __forceinline__ v16h load_a_frag(const _Float16* p, int g) {
  union { v16h v; uint4 q[2]; } u;
  u.q[0] = *(const uint4*)(p + 16 * g);
  u.q[1] = *(const uint4*)(p + 16 * g + 8);
  return u.v;
}

// B-operand (32x16 f16) from row-major BT[n][k]: lane n = lane%16, g = lane/16.
// halves hi = BT[n][16g + hi]  (32B contiguous)
__device__ __forceinline__ v16h load_b_frag(const _Float16* p, int g) {
  union { v16h v; uint4 q[2]; } u;
  u.q[0] = *(const uint4*)(p + 16 * g);
  u.q[1] = *(const uint4*)(p + 16 * g + 8);
  return u.v;
}

// ---------------------------------------------------------------------------
// Tensor Data Mover: 2-D tile load Global -> LDS with per-row LDS padding.
// D# packing per CDNA5 ISA.
// tensor dims == tile dims so no OOB handling is ever exercised.
// This toolchain exposes the 6-arg builtin:
//   (uint32x4 g0, int32x8 g1, int32x4 g2, int32x4 g3, int32x8 g4, i32 cpol)
// ---------------------------------------------------------------------------
__device__ __forceinline__ unsigned lds_addr_of(const void* p) {
  // generic LDS address = {aperture[63:32], lds_byte_offset[31:0]}
  return (unsigned)(uintptr_t)p;
}

__device__ __forceinline__ void tdm_load_2d(unsigned lds_addr,
                                            unsigned long long gaddr,
                                            unsigned tile_d0,    // elements
                                            unsigned tile_d1,    // rows
                                            unsigned stride0,    // elements
                                            unsigned pad_interval_enc,
                                            unsigned pad_amount_enc) {
  v4u g0;
  g0[0] = 1u;                                        // count=1 (valid), user mode
  g0[1] = lds_addr;                                  // lds_addr[31:0]
  g0[2] = (unsigned)gaddr;                           // global_addr[95:64]
  g0[3] = (unsigned)(gaddr >> 32) | (2u << 30);      // addr[56:32] | type=2

  unsigned w0 = (1u << 16)                           // data_size = 2 bytes
              | (1u << 20)                           // pad_enable
              | (pad_interval_enc << 22)
              | (pad_amount_enc << 25);
  v8i g1;
  g1[0] = (int)w0;                                   // wg_mask=0, flags
  g1[1] = (int)((tile_d0 & 0xFFFFu) << 16);          // tensor_dim0[15:0]
  g1[2] = (int)((tile_d0 >> 16) | ((tile_d1 & 0xFFFFu) << 16)); // d0 hi | d1 lo
  g1[3] = (int)((tile_d1 >> 16) | (tile_d0 << 16));  // d1 hi | tile_dim0
  g1[4] = (int)tile_d1;                              // tile_dim1 | tile_dim2=0
  g1[5] = (int)stride0;                              // tensor_dim0_stride[31:0]
  g1[6] = 0;                                         // stride0 hi | stride1 lo
  g1[7] = 0;
  v4i gz4 = {0, 0, 0, 0};
  v8i gz8 = {0, 0, 0, 0, 0, 0, 0, 0};
  __builtin_amdgcn_tensor_load_to_lds(g0, g1, gz4, gz4, gz8, 0);
}

// ---------------------------------------------------------------------------
// Cast kernels
// ---------------------------------------------------------------------------
__global__ void cast_f32_to_f16_kernel(const float* __restrict__ in,
                                       _Float16* __restrict__ out, int n4) {
  int i = blockIdx.x * blockDim.x + threadIdx.x;
  if (i < n4) {
    float4 v = ((const float4*)in)[i];
    union { _Float16 h[4]; uint2 u; } o;
    o.h[0] = (_Float16)v.x; o.h[1] = (_Float16)v.y;
    o.h[2] = (_Float16)v.z; o.h[3] = (_Float16)v.w;
    ((uint2*)out)[i] = o.u;
  }
}

// in: [K,N] f32 row-major -> out: [N,K] f16 row-major (coalesced writes)
__global__ void cast_transpose_kernel(const float* __restrict__ in,
                                      _Float16* __restrict__ out, int K, int N) {
  int idx = blockIdx.x * blockDim.x + threadIdx.x;
  if (idx < K * N) {
    int n = idx / K, k = idx % K;
    out[idx] = (_Float16)in[(size_t)k * N + n];
  }
}

// ---------------------------------------------------------------------------
// Tiled WMMA GEMM with TDM double-buffered tile loads:
//   C[M,N] = A[M,K] * BT[N,K]^T + bias[N]
// Block tile 128x128, BK=64, 256 threads (8 waves), wave tile 32x64.
// Wave 0 drives the Tensor Data Mover; TDM pad feature writes LDT=72 rows.
// ---------------------------------------------------------------------------
template <bool HALF_OUT>
__global__ __launch_bounds__(256) void gemm_wmma_kernel(
    const _Float16* __restrict__ A, const _Float16* __restrict__ BT,
    const float* __restrict__ bias, void* __restrict__ Cout,
    int M, int N, int K) {
  constexpr int BM = 128, BN = 128, BK = 64, LDT = 72;  // 144B rows, 16B aligned
  __shared__ __align__(16) _Float16 As[2][BM * LDT];
  __shared__ __align__(16) _Float16 Bs[2][BN * LDT];

  const int nTiles = N / BN;
  const int m0 = (blockIdx.x / nTiles) * BM;
  const int n0 = (blockIdx.x % nTiles) * BN;
  const int tid = threadIdx.x;
  const int lane = tid & 31, wave = tid >> 5;
  const int g = lane >> 4, ln = lane & 15;
  const int wm = (wave >> 1) * 32;  // 0,32,64,96
  const int wn = (wave & 1) * 64;   // 0,64

  // pad encodings: interval 32 DWORDs (=64 halves) -> 4 ; amount 4 DWORDs -> 3
  const unsigned ldsA[2] = {lds_addr_of(&As[0][0]), lds_addr_of(&As[1][0])};
  const unsigned ldsB[2] = {lds_addr_of(&Bs[0][0]), lds_addr_of(&Bs[1][0])};
  const unsigned long long gA =
      (unsigned long long)(uintptr_t)A + (unsigned long long)m0 * K * 2u;
  const unsigned long long gB =
      (unsigned long long)(uintptr_t)BT + (unsigned long long)n0 * K * 2u;

  v8f zero = {};
  v8f acc[2][4];
  for (int i = 0; i < 2; i++)
    for (int j = 0; j < 4; j++) acc[i][j] = zero;

  const int nIters = K / BK;
  if (wave == 0) {  // prologue: fetch tile 0 into buffer 0
    tdm_load_2d(ldsA[0], gA, BK, BM, (unsigned)K, 4u, 3u);
    tdm_load_2d(ldsB[0], gB, BK, BN, (unsigned)K, 4u, 3u);
  }

  for (int it = 0; it < nIters; ++it) {
    __syncthreads();  // previous compute done -> next buffer is free
    if (wave == 0) {
      if (it + 1 < nIters) {
        unsigned long long off = (unsigned long long)(it + 1) * BK * 2u;
        tdm_load_2d(ldsA[(it + 1) & 1], gA + off, BK, BM, (unsigned)K, 4u, 3u);
        tdm_load_2d(ldsB[(it + 1) & 1], gB + off, BK, BN, (unsigned)K, 4u, 3u);
        __builtin_amdgcn_s_wait_tensorcnt((short)2);  // tile `it` resident
      } else {
        __builtin_amdgcn_s_wait_tensorcnt((short)0);
      }
    }
    __syncthreads();  // tile `it` visible to all waves; TDM it+1 in flight

    const _Float16* Ab = As[it & 1];
    const _Float16* Bb = Bs[it & 1];
    for (int kk = 0; kk < 2; ++kk) {
      v16h af[2], bf[4];
      for (int i = 0; i < 2; i++)
        af[i] = load_a_frag(Ab + (wm + i * 16 + ln) * LDT + kk * 32, g);
      for (int j = 0; j < 4; j++)
        bf[j] = load_b_frag(Bb + (wn + j * 16 + ln) * LDT + kk * 32, g);
      for (int i = 0; i < 2; i++)
        for (int j = 0; j < 4; j++)
          acc[i][j] = wmma32(af[i], bf[j], acc[i][j]);
    }
  }

  for (int i = 0; i < 2; i++)
    for (int j = 0; j < 4; j++) {
      int col = n0 + wn + j * 16 + ln;
      float bc = bias[col];
      for (int r = 0; r < 8; r++) {
        int row = m0 + wm + i * 16 + r + 8 * g;
        float v = acc[i][j][r] + bc;
        if (HALF_OUT)
          ((_Float16*)Cout)[(size_t)row * N + col] = (_Float16)v;
        else
          ((float*)Cout)[(size_t)row * N + col] = v;
      }
    }
}

// ---------------------------------------------------------------------------
// Flash attention: one block per (b, h, 128-row q tile). 8 waves x 16 q rows.
// qkv: [B*T, 6144] f16 (Q | K | V, each [*, h*128+d]); out: [B*T, 2048] f16.
// ---------------------------------------------------------------------------
__global__ __launch_bounds__(256) void attn_wmma_kernel(
    const _Float16* __restrict__ qkv, _Float16* __restrict__ out) {
  constexpr int T = 256, C = 2048, H = 16, Dh = 128, QC = 6144;
  constexpr int LDQ = 136, LDV = 40, LDP = 40;
  __shared__ __align__(16) _Float16 Qs[128 * LDQ];       // 128 q rows x Dh
  __shared__ __align__(16) _Float16 Ks[32 * LDQ];        // 32 keys x Dh
  __shared__ __align__(16) _Float16 Vt[128 * LDV];       // Dh x 32 keys (transposed)
  __shared__ __align__(16) _Float16 Ps[8 * 16 * LDP];    // per-wave 16x32 P scratch

  const int qt = blockIdx.x & 1;
  const int bh = blockIdx.x >> 1;
  const int h = bh % H, b = bh / H;
  const int qBase = qt * 128;
  const _Float16* qPtr = qkv + (size_t)(b * T) * QC + h * Dh;
  const _Float16* kPtr = qPtr + 2048;
  const _Float16* vPtr = qPtr + 4096;

  const int tid = threadIdx.x, lane = tid & 31, wave = tid >> 5;
  const int g = lane >> 4, ln = lane & 15;

  {  // load Q tile: each thread copies 64 contiguous halves
    int row = tid >> 1;
    int c = (tid & 1) * 64;
    const uint4* src = (const uint4*)(qPtr + (size_t)(qBase + row) * QC + c);
    uint4* dst = (uint4*)(Qs + row * LDQ + c);
    for (int i = 0; i < 8; i++) dst[i] = src[i];
  }

  v8f zero = {};
  float mrun[8], lrun[8];
  v8f o[8];
  for (int r = 0; r < 8; r++) { mrun[r] = -1e30f; lrun[r] = 0.f; o[r] = zero; }

  const int myRowMax = qBase + wave * 16 + 15;
  const int nk = (qBase + 128) / 32;
  const float scale = 0.08838834764831845f;  // 1/sqrt(128)

  for (int j = 0; j < nk; j++) {
    const int jBase = j * 32;
    __syncthreads();  // previous tile's compute done before overwrite
    {   // load K tile + V tile (V transposed into Vt[dh][key])
      int krow = tid >> 3;          // 0..31
      int c = (tid & 7) * 16;       // Dh chunk
      const uint4* ksrc = (const uint4*)(kPtr + (size_t)(jBase + krow) * QC + c);
      uint4* kdst = (uint4*)(Ks + krow * LDQ + c);
      kdst[0] = ksrc[0]; kdst[1] = ksrc[1];
      union { uint4 q[2]; _Float16 hh[16]; } vv;
      const uint4* vsrc = (const uint4*)(vPtr + (size_t)(jBase + krow) * QC + c);
      vv.q[0] = vsrc[0]; vv.q[1] = vsrc[1];
      for (int i = 0; i < 16; i++) Vt[(c + i) * LDV + krow] = vv.hh[i];
    }
    __syncthreads();

    if (jBase <= myRowMax) {  // wave-uniform: EXEC stays all-ones
      // S = Q * K^T : two 16-key subtiles, K-dim = Dh in 4 steps of 32
      v8f s[2]; s[0] = zero; s[1] = zero;
      for (int kk = 0; kk < 4; kk++) {
        v16h aq = load_a_frag(Qs + (wave * 16 + ln) * LDQ + kk * 32, g);
        for (int sn = 0; sn < 2; sn++) {
          v16h bk = load_b_frag(Ks + (sn * 16 + ln) * LDQ + kk * 32, g);
          s[sn] = wmma32(aq, bk, s[sn]);
        }
      }
      // causal mask + scale + row max (row r+8g lives in one 16-lane half)
      float mt[8];
      for (int r = 0; r < 8; r++) {
        int row = qBase + wave * 16 + r + 8 * g;
        float s0 = (jBase + ln      <= row) ? s[0][r] * scale : -1e30f;
        float s1 = (jBase + 16 + ln <= row) ? s[1][r] * scale : -1e30f;
        s[0][r] = s0; s[1][r] = s1;
        mt[r] = fmaxf(s0, s1);
      }
      for (int off = 8; off >= 1; off >>= 1)
        for (int r = 0; r < 8; r++)
          mt[r] = fmaxf(mt[r], __shfl_xor(mt[r], off));
      float corr[8];
      for (int r = 0; r < 8; r++) {
        float mnew = fmaxf(mrun[r], mt[r]);
        corr[r] = __expf(mrun[r] - mnew);
        mrun[r] = mnew;
      }
      float rs[8];
      for (int r = 0; r < 8; r++) {
        float p0 = __expf(s[0][r] - mrun[r]);
        float p1 = __expf(s[1][r] - mrun[r]);
        s[0][r] = p0; s[1][r] = p1;
        rs[r] = p0 + p1;
      }
      for (int off = 8; off >= 1; off >>= 1)
        for (int r = 0; r < 8; r++)
          rs[r] += __shfl_xor(rs[r], off);
      for (int r = 0; r < 8; r++) lrun[r] = lrun[r] * corr[r] + rs[r];
      for (int d = 0; d < 8; d++)
        for (int r = 0; r < 8; r++) o[d][r] *= corr[r];

      // P: C-layout -> A-layout via per-wave LDS scratch (same-wave, in-order)
      _Float16* pw = Ps + wave * 16 * LDP;
      for (int r = 0; r < 8; r++) {
        pw[(r + 8 * g) * LDP + ln]      = (_Float16)s[0][r];
        pw[(r + 8 * g) * LDP + 16 + ln] = (_Float16)s[1][r];
      }
      v16h pf = load_a_frag(pw + ln * LDP, g);
      for (int d = 0; d < 8; d++) {
        v16h vf = load_b_frag(Vt + (d * 16 + ln) * LDV, g);
        o[d] = wmma32(pf, vf, o[d]);
      }
    }
  }

  // normalize + store (head-concatenated [B,T,C] f16)
  for (int r = 0; r < 8; r++) {
    int row = qBase + wave * 16 + r + 8 * g;
    float inv = 1.0f / lrun[r];
    size_t base = ((size_t)(b * T + row)) * C + h * Dh + ln;
    for (int d = 0; d < 8; d++)
      out[base + d * 16] = (_Float16)(o[d][r] * inv);
  }
}

// ---------------------------------------------------------------------------
extern "C" void kernel_launch(void* const* d_in, const int* in_sizes, int n_in,
                              void* d_out, int out_size, void* d_ws,
                              size_t ws_size, hipStream_t stream) {
  (void)in_sizes; (void)n_in; (void)out_size; (void)ws_size;
  const float* x      = (const float*)d_in[0];
  const float* qkv_w  = (const float*)d_in[1];
  const float* qkv_b  = (const float*)d_in[2];
  const float* out_w  = (const float*)d_in[3];
  const float* out_b  = (const float*)d_in[4];
  float* out = (float*)d_out;

  constexpr int B = 64, T = 256, C = 2048, H = 16, QC = 3 * C;
  constexpr size_t M = (size_t)B * T;  // 16384

  char* w = (char*)d_ws;
  _Float16* xh    = (_Float16*)w;  w += M * C * 2;             // 64 MB
  _Float16* wqkvT = (_Float16*)w;  w += (size_t)QC * C * 2;    // 24 MB
  _Float16* woutT = (_Float16*)w;  w += (size_t)C * C * 2;     //  8 MB
  _Float16* qkvh  = (_Float16*)w;  w += M * QC * 2;            // 192 MB
  _Float16* attn  = (_Float16*)w;                              // 64 MB

  int n4 = (int)(M * C / 4);
  cast_f32_to_f16_kernel<<<(n4 + 255) / 256, 256, 0, stream>>>(x, xh, n4);
  int nq = C * QC;
  cast_transpose_kernel<<<(nq + 255) / 256, 256, 0, stream>>>(qkv_w, wqkvT, C, QC);
  int no = C * C;
  cast_transpose_kernel<<<(no + 255) / 256, 256, 0, stream>>>(out_w, woutT, C, C);

  // qkv = xh @ qkv_w + qkv_b   (f16 out)
  gemm_wmma_kernel<true><<<(int)(M / 128) * (QC / 128), 256, 0, stream>>>(
      xh, wqkvT, qkv_b, qkvh, (int)M, QC, C);

  // attention
  attn_wmma_kernel<<<B * H * 2, 256, 0, stream>>>(qkvh, attn);

  // out = attn @ out_w + out_b   (f32 out)
  gemm_wmma_kernel<false><<<(int)(M / 128) * (C / 128), 256, 0, stream>>>(
      attn, woutT, out_b, out, (int)M, C, C);
}
